// MultiHeadAttention_137438954246
// MI455X (gfx1250) — matrix operand order repaired and checked by reading, not compiled, not run on hardware
//
#include <hip/hip_runtime.h>
#include <hip/hip_bf16.h>

// ---------------------------------------------------------------------------
// MHA forward for gfx1250 (CDNA5, wave32, WMMA).
// All matmuls (projections, Q*K^T, P*V, softmax row-sums) run on
// v_wmma_f32_16x16x32_bf16 with f32 accumulation. Attention K/V tiles are
// staged with global_load_async_to_lds_b128 (double-buffered, ASYNCcnt).
// ---------------------------------------------------------------------------

typedef __bf16        bf16x16 __attribute__((ext_vector_type(16)));
typedef float         f32x8   __attribute__((ext_vector_type(8)));
typedef float         f32x4   __attribute__((ext_vector_type(4)));
typedef unsigned int  u32x4v  __attribute__((ext_vector_type(4)));
typedef unsigned int  u32x2v  __attribute__((ext_vector_type(2)));
typedef int           i32x4g  __attribute__((vector_size(16)));   // builtin's pointee type

#define D_MODEL 1024
#define SEQ     2048
#define BATCH   2
#define HEADS   16
#define DK      64
#define MTOT    (BATCH * SEQ)   // 4096

#if __has_builtin(__builtin_amdgcn_global_load_async_to_lds_b128)
#define HAVE_ASYNC_LDS 1
#else
#define HAVE_ASYNC_LDS 0
#endif

__device__ __forceinline__ unsigned short f2bf(float x) {
    unsigned u = __float_as_uint(x);
    unsigned r = u + 0x7FFFu + ((u >> 16) & 1u);   // round-to-nearest-even
    return (unsigned short)(r >> 16);
}

__device__ __forceinline__ f32x8 zero8() {
    f32x8 z;
#pragma unroll
    for (int i = 0; i < 8; ++i) z[i] = 0.0f;
    return z;
}

union FragU { bf16x16 v; u32x4v q[2]; };

// 32-byte WMMA fragment as two 16B pieces (ds_load_b128 / global_load_b128).
__device__ __forceinline__ bf16x16 ld_frag(const unsigned short* p0,
                                           const unsigned short* p1) {
    FragU f;
    f.q[0] = *(const u32x4v*)p0;
    f.q[1] = *(const u32x4v*)p1;
    return f.v;
}

__device__ __forceinline__ f32x8 wmma_bf16(bf16x16 a, bf16x16 b, f32x8 c) {
    return __builtin_amdgcn_wmma_f32_16x16x32_bf16(
        /*neg_a=*/false, a, /*neg_b=*/false, b,
        /*c_mod=*/(short)0, c, /*reuse_a=*/false, /*reuse_b=*/false);
}

// ---- 16-lane xor-butterfly max: v_permlane16 (VALU, co-executes with WMMA)
// instead of ds_bpermute (would contend with WMMA fragment loads on LDS).
#if __has_builtin(__builtin_amdgcn_permlane16)
__device__ __forceinline__ float plx16(float v, unsigned s0, unsigned s1) {
    unsigned x = (unsigned)__float_as_int(v);
    unsigned y = __builtin_amdgcn_permlane16(x, x, s0, s1, false, false);
    return __int_as_float((int)y);
}
__device__ __forceinline__ float rowmax16(float v) {
    v = fmaxf(v, plx16(v, 0x67452301u, 0xEFCDAB89u));   // lane ^ 1
    v = fmaxf(v, plx16(v, 0x54761032u, 0xDCFE98BAu));   // lane ^ 2
    v = fmaxf(v, plx16(v, 0x32107654u, 0xBA98FEDCu));   // lane ^ 4
    v = fmaxf(v, plx16(v, 0xFEDCBA98u, 0x76543210u));   // lane ^ 8
    return v;
}
#else
__device__ __forceinline__ float rowmax16(float v) {
#pragma unroll
    for (int d = 1; d < 16; d <<= 1) v = fmaxf(v, __shfl_xor(v, d));
    return v;
}
#endif

// ---- 16B global -> LDS copy: async (ASYNCcnt, no VGPR data) when available.
__device__ __forceinline__ void cp16_g2l(const unsigned short* g, unsigned short* l) {
#if HAVE_ASYNC_LDS
    __builtin_amdgcn_global_load_async_to_lds_b128(
        (__attribute__((address_space(1))) i32x4g*)g,
        (__attribute__((address_space(3))) i32x4g*)l,
        /*offset=*/0, /*cpol=*/0);
#else
    *(u32x4v*)l = *(const u32x4v*)g;
#endif
}

__device__ __forceinline__ void wait_async_lds() {
#if HAVE_ASYNC_LDS
#if __has_builtin(__builtin_amdgcn_s_wait_asynccnt)
    __builtin_amdgcn_s_wait_asynccnt(0);
#else
    asm volatile("s_wait_asynccnt 0x0" ::: "memory");
#endif
#endif
}

// ---------------------------------------------------------------------------
// GEMM:  OUT = A[4096,1024] * W[1024,1024]^T + bias
//   mode 0/1 : bf16 -> [B,H,S,64]   (Q / K layout)
//   mode 2   : bf16 -> [B,H,64,S]   (V transposed for the P*V B-operand)
//   mode 3   : fp32 -> [4096,1024]  (final output)
// Block tile 128x128, K-step 32; 8 waves, each 64x32 (4x2 WMMA tiles).
// ---------------------------------------------------------------------------
template <bool A_IS_BF16>
__global__ __launch_bounds__(256) void gemm_xwT_kernel(
    const void*  __restrict__ Aptr,
    const float* __restrict__ W,
    const float* __restrict__ bias,
    void*        __restrict__ Out,
    int mode)
{
    constexpr int LDT = 48;                 // LDS row stride (halves) = 96B
    __shared__ alignas(16) unsigned short lsA[128 * LDT];
    __shared__ alignas(16) unsigned short lsW[128 * LDT];

    const int tid  = threadIdx.x;
    const int wid  = tid >> 5;
    const int lane = tid & 31;
    const int half = lane >> 4;
    const int lr   = lane & 15;
    const int mBase = blockIdx.y * 128;
    const int nBase = blockIdx.x * 128;
    const int wm = (wid >> 2) * 64;
    const int wn = (wid & 3) * 32;

    f32x8 acc[4][2];
#pragma unroll
    for (int mt = 0; mt < 4; ++mt)
#pragma unroll
        for (int nt = 0; nt < 2; ++nt) acc[mt][nt] = zero8();

    for (int k0 = 0; k0 < D_MODEL; k0 += 32) {
        // ---- stage W tile 128x32 (fp32 -> bf16) -------------------------
#pragma unroll
        for (int i = 0; i < 4; ++i) {
            int idx = tid + i * 256;
            int row = idx >> 3;
            int c4  = (idx & 7) << 2;
            f32x4 v = *(const f32x4*)(W + (size_t)(nBase + row) * D_MODEL + k0 + c4);
            u32x2v p;
            p.x = (unsigned)f2bf(v.x) | ((unsigned)f2bf(v.y) << 16);
            p.y = (unsigned)f2bf(v.z) | ((unsigned)f2bf(v.w) << 16);
            *(u32x2v*)&lsW[row * LDT + c4] = p;
        }
        // ---- stage A tile 128x32 ----------------------------------------
        if constexpr (A_IS_BF16) {
            const unsigned short* A = (const unsigned short*)Aptr;
#pragma unroll
            for (int i = 0; i < 2; ++i) {
                int idx = tid + i * 256;
                int row = idx >> 2;
                int ch  = (idx & 3) << 3;
                *(u32x4v*)&lsA[row * LDT + ch] =
                    *(const u32x4v*)(A + (size_t)(mBase + row) * D_MODEL + k0 + ch);
            }
        } else {
            const float* A = (const float*)Aptr;
#pragma unroll
            for (int i = 0; i < 4; ++i) {
                int idx = tid + i * 256;
                int row = idx >> 3;
                int c4  = (idx & 7) << 2;
                f32x4 v = *(const f32x4*)(A + (size_t)(mBase + row) * D_MODEL + k0 + c4);
                u32x2v p;
                p.x = (unsigned)f2bf(v.x) | ((unsigned)f2bf(v.y) << 16);
                p.y = (unsigned)f2bf(v.z) | ((unsigned)f2bf(v.w) << 16);
                *(u32x2v*)&lsA[row * LDT + c4] = p;
            }
        }
        if (k0 + 32 < D_MODEL) {   // hint next tiles toward cache
            __builtin_prefetch(W + (size_t)(nBase + (tid >> 3)) * D_MODEL + k0 + 32, 0, 0);
            __builtin_prefetch((const char*)Aptr +
                               ((size_t)(mBase + (tid >> 3)) * D_MODEL + k0 + 32) *
                                   (A_IS_BF16 ? 2 : 4), 0, 0);
        }
        __syncthreads();

        // B-operand frags: lane = col, K = half*16 .. +15
        bf16x16 bf[2];
#pragma unroll
        for (int nt = 0; nt < 2; ++nt) {
            const unsigned short* p = &lsW[(wn + nt * 16 + lr) * LDT + half * 16];
            bf[nt] = ld_frag(p, p + 8);
        }
        // A-operand frags: lane = row, K = half*16 .. +15
#pragma unroll
        for (int mt = 0; mt < 4; ++mt) {
            const unsigned short* p = &lsA[(wm + mt * 16 + lr) * LDT + half * 16];
            bf16x16 af = ld_frag(p, p + 8);
#pragma unroll
            for (int nt = 0; nt < 2; ++nt)
                acc[mt][nt] = wmma_bf16(af, bf[nt], acc[mt][nt]);
        }
        __syncthreads();
    }

    // ---- epilogue: C layout -> lane holds (M = r + 8*half, N = lr) ------
#pragma unroll
    for (int mt = 0; mt < 4; ++mt)
#pragma unroll
        for (int nt = 0; nt < 2; ++nt) {
            int gn = nBase + wn + nt * 16 + lr;
            float bv = bias[gn];
#pragma unroll
            for (int r = 0; r < 8; ++r) {
                int gm = mBase + wm + mt * 16 + r + half * 8;
                float val = acc[mt][nt][r] + bv;
                if (mode == 3) {
                    ((float*)Out)[(size_t)gm * D_MODEL + gn] = val;
                } else {
                    int bb = gm >> 11, st = gm & (SEQ - 1);
                    int hh = gn >> 6,  dd = gn & (DK - 1);
                    unsigned short o = f2bf(val);
                    if (mode == 2)   // V transposed: [B,H,64,S]
                        ((unsigned short*)Out)[(((size_t)bb * HEADS + hh) * DK + dd) * SEQ + st] = o;
                    else             // Q/K: [B,H,S,64]
                        ((unsigned short*)Out)[(((size_t)bb * HEADS + hh) * SEQ + st) * DK + dd] = o;
                }
            }
        }
}

// ---------------------------------------------------------------------------
// Flash attention: block = one (b,h) x 128 queries; wave = 16 queries.
// Key loop (step 32): double-buffered async K/V staging; scores = Q*K^T
// (4 WMMAs); online softmax with permlane16 row-max; P -> per-wave LDS;
// row-sums as one extra WMMA against a ones-matrix; ctx += P*V (4 WMMAs).
// ---------------------------------------------------------------------------
__global__ __launch_bounds__(256) void attn_kernel(
    const unsigned short* __restrict__ Qh,
    const unsigned short* __restrict__ Kh,
    const unsigned short* __restrict__ Vt,
    unsigned short*       __restrict__ Ctx)
{
    constexpr int LK = 72;   // [32 keys][64 d]  row stride 144B
    constexpr int LV = 48;   // [64 d][32 keys]  row stride 96B
    constexpr int LP = 48;   // per-wave P [16][32]
    __shared__ alignas(16) unsigned short lsK[2][32 * LK];
    __shared__ alignas(16) unsigned short lsV[2][64 * LV];
    __shared__ alignas(16) unsigned short lsP[8][16 * LP];

    const int tid  = threadIdx.x;
    const int wid  = tid >> 5;
    const int lane = tid & 31;
    const int half = lane >> 4;
    const int lr   = lane & 15;
    const int bh   = blockIdx.y;               // b*16 + h
    const int b    = bh >> 4, h = bh & 15;
    const unsigned short* Qb = Qh + (size_t)bh * SEQ * DK;
    const unsigned short* Kb = Kh + (size_t)bh * SEQ * DK;
    const unsigned short* Vb = Vt + (size_t)bh * DK * SEQ;
    const int q0 = blockIdx.x * 128 + wid * 16;

    // per-thread slice coordinates for cooperative tile staging
    const int rowK = tid >> 3, chK = (tid & 7) << 3;   // K tile: 32 x 64 halves
    const int rowV = tid >> 2, chV = (tid & 3) << 3;   // V tile: 64 x 32 halves

    // Q fragments (A-operand, d split into two K=32 chunks), live for all keys
    bf16x16 qf[2];
#pragma unroll
    for (int c = 0; c < 2; ++c) {
        const unsigned short* p = Qb + (size_t)(q0 + lr) * DK + c * 32 + half * 16;
        qf[c] = ld_frag(p, p + 8);
    }

    // ones B-operand (bf16 1.0 everywhere) for WMMA row-sums of P
    FragU onesu;
    onesu.q[0] = (u32x4v){0x3F803F80u, 0x3F803F80u, 0x3F803F80u, 0x3F803F80u};
    onesu.q[1] = onesu.q[0];
    const bf16x16 onesf = onesu.v;

    float mrow[8], lrow[8];
    f32x8 o[4];
#pragma unroll
    for (int r = 0; r < 8; ++r) { mrow[r] = -3.0e38f; lrow[r] = 0.0f; }
#pragma unroll
    for (int et = 0; et < 4; ++et) o[et] = zero8();

    // prologue: stage tile 0
    cp16_g2l(Kb + (size_t)rowK * DK + chK,  &lsK[0][rowK * LK + chK]);
    cp16_g2l(Vb + (size_t)rowV * SEQ + chV, &lsV[0][rowV * LV + chV]);
    wait_async_lds();
    __syncthreads();

    for (int it = 0; it < SEQ / 32; ++it) {
        const int j0  = it * 32;
        const int buf = it & 1;
        // prefetch next tile into the other buffer; overlaps with compute
        if (it + 1 < SEQ / 32) {
            cp16_g2l(Kb + (size_t)(j0 + 32 + rowK) * DK + chK,
                     &lsK[buf ^ 1][rowK * LK + chK]);
            cp16_g2l(Vb + (size_t)rowV * SEQ + (j0 + 32) + chV,
                     &lsV[buf ^ 1][rowV * LV + chV]);
        }

        // ---- scores: S[16x32] = Q(16x64) * K^T(64x32) -------------------
        f32x8 s[2];
#pragma unroll
        for (int nt = 0; nt < 2; ++nt) {
            const unsigned short* p0 = &lsK[buf][(nt * 16 + lr) * LK + half * 16];
            const unsigned short* p1 = &lsK[buf][(nt * 16 + lr) * LK + 32 + half * 16];
            bf16x16 kf0 = ld_frag(p0, p0 + 8);
            bf16x16 kf1 = ld_frag(p1, p1 + 8);
            f32x8 sc = zero8();
            sc = wmma_bf16(qf[0], kf0, sc);
            sc = wmma_bf16(qf[1], kf1, sc);
            s[nt] = sc;
        }

        // ---- online softmax (row r+8*half spans one 16-lane half) -------
        float alpha8[8];
#pragma unroll
        for (int r = 0; r < 8; ++r) {
            float a0 = s[0][r] * 0.125f;           // 1/sqrt(64)
            float a1 = s[1][r] * 0.125f;
            float mx = rowmax16(fmaxf(a0, a1));
            float mnew = fmaxf(mrow[r], mx);
            alpha8[r] = __expf(mrow[r] - mnew);
            mrow[r] = mnew;
            float p0 = __expf(a0 - mnew);
            float p1 = __expf(a1 - mnew);
#pragma unroll
            for (int et = 0; et < 4; ++et) o[et][r] *= alpha8[r];
            int m = r + half * 8;                  // C-layout row
            lsP[wid][m * LP +      lr] = f2bf(p0);
            lsP[wid][m * LP + 16 + lr] = f2bf(p1);
        }
        // per-wave LDS region; same-wave DS ops stay in order -> no barrier

        // P back as A-operand; row-sums via WMMA against ones (broadcasts
        // each row's sum to all 16 lanes of its half -> no shuffle needed)
        const unsigned short* pp = &lsP[wid][lr * LP + half * 16];
        bf16x16 pf = ld_frag(pp, pp + 8);
        f32x8 rs = wmma_bf16(pf, onesf, zero8());
#pragma unroll
        for (int r = 0; r < 8; ++r) lrow[r] = lrow[r] * alpha8[r] + rs[r];

        // ---- ctx += P(16x32) * V(32x64) ---------------------------------
#pragma unroll
        for (int et = 0; et < 4; ++et) {
            const unsigned short* pv = &lsV[buf][(et * 16 + lr) * LV + half * 16];
            bf16x16 vf = ld_frag(pv, pv + 8);
            o[et] = wmma_bf16(pf, vf, o[et]);
        }

        wait_async_lds();      // next tile landed
        __syncthreads();       // everyone done with buf, buf^1 visible
    }

    // ---- finalize: /= rowsum, write ctx bf16 as [B,S,H*64] --------------
#pragma unroll
    for (int r = 0; r < 8; ++r) {
        float inv = 1.0f / lrow[r];
        int tok = q0 + r + half * 8;
#pragma unroll
        for (int et = 0; et < 4; ++et) {
            int ch = h * DK + et * 16 + lr;
            Ctx[((size_t)b * SEQ + tok) * D_MODEL + ch] = f2bf(o[et][r] * inv);
        }
    }
}

// ---------------------------------------------------------------------------
// Launch: 3 projection GEMMs -> flash attention -> output GEMM
// Workspace (bf16): Qh[2,16,2048,64] Kh[same] Vt[2,16,64,2048] Ctx[2,2048,1024]
// = 4 x 8 MiB = 32 MiB.
// ---------------------------------------------------------------------------
extern "C" void kernel_launch(void* const* d_in, const int* in_sizes, int n_in,
                              void* d_out, int out_size, void* d_ws, size_t ws_size,
                              hipStream_t stream) {
    (void)in_sizes; (void)n_in; (void)out_size; (void)ws_size;
    const float* query = (const float*)d_in[0];
    const float* key   = (const float*)d_in[1];
    const float* value = (const float*)d_in[2];
    const float* w_q   = (const float*)d_in[3];
    const float* b_q   = (const float*)d_in[4];
    const float* w_k   = (const float*)d_in[5];
    const float* b_k   = (const float*)d_in[6];
    const float* w_v   = (const float*)d_in[7];
    const float* b_v   = (const float*)d_in[8];
    const float* w_o   = (const float*)d_in[9];
    const float* b_o   = (const float*)d_in[10];

    const size_t QKV_ELEMS = (size_t)BATCH * HEADS * SEQ * DK;   // 4 Mi halves
    unsigned short* Qh  = (unsigned short*)d_ws;
    unsigned short* Kh  = Qh + QKV_ELEMS;
    unsigned short* Vt  = Kh + QKV_ELEMS;
    unsigned short* Ctx = Vt + QKV_ELEMS;

    dim3 gg(D_MODEL / 128, MTOT / 128);   // (8, 32)
    dim3 bb(256);
    gemm_xwT_kernel<false><<<gg, bb, 0, stream>>>(query, w_q, b_q, Qh, 0);
    gemm_xwT_kernel<false><<<gg, bb, 0, stream>>>(key,   w_k, b_k, Kh, 1);
    gemm_xwT_kernel<false><<<gg, bb, 0, stream>>>(value, w_v, b_v, Vt, 2);

    attn_kernel<<<dim3(SEQ / 128, BATCH * HEADS), bb, 0, stream>>>(Qh, Kh, Vt, Ctx);

    gemm_xwT_kernel<true><<<gg, bb, 0, stream>>>(Ctx, w_o, b_o, d_out, 3);
}
